// Mamba3Block_6674379178231
// MI455X (gfx1250) — matrix operand order repaired and checked by reading, not compiled, not run on hardware
//
#include <hip/hip_runtime.h>
#include <hip/hip_bf16.h>

// ---------------- model dims ----------------
#define DMODEL   4096
#define DINNER   8192
#define NH       128
#define HD       64
#define NG       8
#define DSTATE   128
#define GN       (NG*DSTATE)        // 1024
#define CONVDIM  (DINNER + 2*GN)    // 10240
#define DPROJ    (2*DINNER + 2*GN + NH) // 18560
#define SEQ      2048
#define CHNK     256
#define NCH      (SEQ/CHNK)         // 8

typedef __attribute__((ext_vector_type(16))) __bf16 bf16x16;
typedef __attribute__((ext_vector_type(8)))  float  f32x8;

union FragB { bf16x16 v; unsigned short u[16]; uint4 q[2]; };
union AccF  { f32x8 v; float f[8]; };
union Q8    { uint4 q; unsigned short u[8]; };

static __device__ __forceinline__ unsigned short f2bf(float f) {
    union { float f; unsigned u; } v; v.f = f;
    unsigned r = v.u + 0x7FFFu + ((v.u >> 16) & 1u);
    return (unsigned short)(r >> 16);
}
static __device__ __forceinline__ float bf2f(unsigned short h) {
    union { unsigned u; float f; } v; v.u = ((unsigned)h) << 16;
    return v.f;
}
static __device__ __forceinline__ bf16x16 ld_frag(const unsigned short* p0,
                                                  const unsigned short* p1) {
    FragB f;
    f.q[0] = *(const uint4*)p0;
    f.q[1] = *(const uint4*)p1;
    return f.v;
}
static __device__ __forceinline__ f32x8 wmma_bf(bf16x16 a, bf16x16 b, f32x8 c) {
    return __builtin_amdgcn_wmma_f32_16x16x32_bf16(false, a, false, b,
                                                   (short)0, c, false, false);
}
static __device__ __forceinline__ float siluf(float x) {
    return x / (1.0f + __expf(-x));
}

// async 32B global -> LDS copy (two b128 DMA issues, ASYNCcnt-tracked)
#define ASYNC_LD32(ldsoff, gaddr)                                              \
    asm volatile("global_load_async_to_lds_b128 %0, %1, off\n\t"               \
                 "global_load_async_to_lds_b128 %0, %1, off offset:16"         \
                 :: "v"(ldsoff), "v"(gaddr) : "memory")

// ---------------- cast f32 -> bf16 ----------------
__global__ void castK(const float* __restrict__ src, unsigned short* __restrict__ dst,
                      size_t n) {
    for (size_t i = (size_t)blockIdx.x * blockDim.x + threadIdx.x; i < n;
         i += (size_t)gridDim.x * blockDim.x)
        dst[i] = f2bf(src[i]);
}

// ---------------- WMMA GEMM: C[M,N] = A[M,K] * W[N,K]^T (+resid) ----------------
// A,W bf16 row-major. M%128==0, N%128==0, K%32==0. 256 threads, 8 waves.
// Double-buffered LDS staging via GLOBAL_LOAD_ASYNC_TO_LDS_B128 (ASYNCcnt DMA);
// steady-state loop prefetches unconditionally, last tile peeled.
__global__ void gemmK(const unsigned short* __restrict__ A,
                      const unsigned short* __restrict__ W,
                      float* __restrict__ C, const float* __restrict__ resid,
                      int M, int N, int K) {
    constexpr int LB = 40; // u16 stride (80B rows, 16B-aligned fragments)
    __shared__ unsigned short sA[2][128 * LB];
    __shared__ unsigned short sW[2][128 * LB];
    const int tid = threadIdx.x, lane = tid & 31, wid = tid >> 5;
    const int wm = (wid & 3) * 32;   // wave row in tile
    const int wn = (wid >> 2) * 64;  // wave col in tile
    const int bm = blockIdx.y * 128, bn = blockIdx.x * 128;
    const int lr = tid & 127, lseg = (tid >> 7) * 16;
    const int ak = ((lane >> 4) * 8);   // A fragment k base (u16)
    const int bk = ((lane >> 4) * 8);   // B fragment k base (u16)
    const int am = lane & 15;

    // per-lane global byte addresses (advance 64B per K-tile)
    unsigned long long ga = (unsigned long long)(A + (size_t)(bm + lr) * K + lseg);
    unsigned long long gw = (unsigned long long)(W + (size_t)(bn + lr) * K + lseg);
    // per-lane LDS byte offsets (flat->LDS truncates to addr[31:0])
    unsigned ldsAb[2], ldsWb[2];
    ldsAb[0] = (unsigned)(unsigned long long)&sA[0][lr * LB + lseg];
    ldsAb[1] = (unsigned)(unsigned long long)&sA[1][lr * LB + lseg];
    ldsWb[0] = (unsigned)(unsigned long long)&sW[0][lr * LB + lseg];
    ldsWb[1] = (unsigned)(unsigned long long)&sW[1][lr * LB + lseg];

    AccF acc[2][4] = {};

    auto compute = [&](const unsigned short* bA, const unsigned short* bW) {
#pragma unroll
        for (int mt = 0; mt < 2; ++mt) {
            const unsigned short* ra = &bA[(wm + mt * 16 + am) * LB + ak];
            bf16x16 af = ld_frag(ra, ra + 16);
#pragma unroll
            for (int nt = 0; nt < 4; ++nt) {
                const unsigned short* rb = &bW[(wn + nt * 16 + am) * LB + bk];
                bf16x16 bf = ld_frag(rb, rb + 16);
                acc[mt][nt].v = wmma_bf(af, bf, acc[mt][nt].v);
            }
        }
    };

    // prologue: tile 0 -> buffer 0
    ASYNC_LD32(ldsAb[0], ga);
    ASYNC_LD32(ldsWb[0], gw);

    const int T = K / 32;
    for (int i = 0; i < T - 1; ++i) {
        const int cur = i & 1;
        ga += 64; gw += 64;
        ASYNC_LD32(ldsAb[cur ^ 1], ga);   // prefetch tile i+1
        ASYNC_LD32(ldsWb[cur ^ 1], gw);
        // 4 newer DMAs outstanding; <=4 means current tile has landed
        asm volatile("s_wait_asynccnt 0x4" ::: "memory");
        __syncthreads(); // all waves' current tile visible
        compute(sA[cur], sW[cur]);
        __syncthreads(); // everyone done reading buf[cur] before it is re-filled
    }
    // tail: last tile
    asm volatile("s_wait_asynccnt 0x0" ::: "memory");
    __syncthreads();
    compute(sA[(T - 1) & 1], sW[(T - 1) & 1]);

#pragma unroll
    for (int mt = 0; mt < 2; ++mt)
#pragma unroll
        for (int nt = 0; nt < 4; ++nt)
#pragma unroll
            for (int r = 0; r < 8; ++r) {
                int row = bm + wm + mt * 16 + ((lane >> 4) * 8) + r;
                int col = bn + wn + nt * 16 + (lane & 15);
                float v = acc[mt][nt].f[r];
                if (resid) v += resid[(size_t)row * N + col];
                C[(size_t)row * N + col] = v;
            }
}

// ---------------- dt = softplus(dt_raw + bias), dA = dt * A ----------------
__global__ void dtK(const float* __restrict__ zx, const float* __restrict__ dt_bias,
                    const float* __restrict__ A_log, float* __restrict__ dt,
                    float* __restrict__ dA) {
    int idx = blockIdx.x * 256 + threadIdx.x; // l*NH + h
    int l = idx >> 7, h = idx & 127;
    float raw = zx[(size_t)l * DPROJ + DINNER + CONVDIM + h] + dt_bias[h];
    float d = (raw > 20.0f) ? raw : log1pf(__expf(raw));
    dt[idx] = d;
    dA[idx] = d * (-__expf(A_log[h]));
}

// ---------------- causal conv(K=4) + silu, split x/B/C, build xdt(bf16) ----------------
__global__ void convK_(const float* __restrict__ zx, const float* __restrict__ cw,
                       const float* __restrict__ cb, const float* __restrict__ dt,
                       float* __restrict__ x_f32, unsigned short* __restrict__ xdt_bf,
                       unsigned short* __restrict__ Bbf, unsigned short* __restrict__ Cbf) {
    int idx = blockIdx.x * 256 + threadIdx.x; // l*CONVDIM + ch
    int l = idx / CONVDIM, ch = idx - l * CONVDIM;
    float acc = cb[ch];
#pragma unroll
    for (int k = 0; k < 4; ++k) {
        int t = l - 3 + k;
        if (t >= 0)
            acc += zx[(size_t)t * DPROJ + DINNER + ch] * cw[ch * 4 + k];
    }
    float s = siluf(acc);
    if (ch < DINNER) {
        x_f32[(size_t)l * DINNER + ch] = s;
        int h = ch >> 6;
        xdt_bf[(size_t)l * DINNER + ch] = f2bf(s * dt[l * NH + h]);
    } else if (ch < DINNER + GN) {
        Bbf[(size_t)l * GN + (ch - DINNER)] = f2bf(s);
    } else {
        Cbf[(size_t)l * GN + (ch - DINNER - GN)] = f2bf(s);
    }
}

// ---------------- per-(chunk,head) inclusive cumsum of dA ----------------
__global__ void cumsumK(const float* __restrict__ dA, float* __restrict__ Acs,
                        float* __restrict__ dAsum) {
    int b = blockIdx.x; // c*NH + h
    int c = b >> 7, h = b & 127, t = threadIdx.x;
    __shared__ float s[256];
    s[t] = dA[(size_t)(c * CHNK + t) * NH + h];
    __syncthreads();
    for (int off = 1; off < 256; off <<= 1) {
        float v = (t >= off) ? s[t - off] : 0.0f;
        __syncthreads();
        s[t] += v;
        __syncthreads();
    }
    Acs[(size_t)b * CHNK + t] = s[t];
    if (t == 255) dAsum[b] = s[255];
}

// ---------------- SSD chunk scan: Y_diag + per-chunk states (WMMA) ----------------
// one block per (chunk,head); 256 threads = 8 waves; dynamic LDS (~285KB, fits 320KB WGP).
#define S_LB  136  // stride for sB/sC rows (256 x 128)
#define S_LT  264  // stride for transposed tiles (rows of length 256)
#define S_LG  40   // per-wave G staging stride
__global__ void scanK(const unsigned short* __restrict__ Bbf,
                      const unsigned short* __restrict__ Cbf,
                      const unsigned short* __restrict__ xdt_bf,
                      const float* __restrict__ Acs,
                      float* __restrict__ Ybuf, float* __restrict__ statesBuf) {
    extern __shared__ char smem[];
    unsigned short* sB    = (unsigned short*)smem;              // [256][S_LB]
    unsigned short* sC    = sB    + 256 * S_LB;                 // [256][S_LB]
    unsigned short* sBT   = sC    + 256 * S_LB;                 // [128][S_LT]  B^T
    unsigned short* sXdT  = sBT   + 128 * S_LT;                 // [64][S_LT]   xdt^T
    unsigned short* sXddT = sXdT  + 64 * S_LT;                  // [64][S_LT]   (xdt*decay)^T
    unsigned short* sG    = sXddT + 64 * S_LT;                  // 8 waves x [16][S_LG]
    float*          scs   = (float*)(sG + 8 * 16 * S_LG);       // [256]

    const int blk = blockIdx.x, c = blk >> 7, h = blk & 127;
    const int tid = threadIdx.x, lane = tid & 31, wid = tid >> 5;
    const int grp = h >> 4; // head group (rep = 16)

    // ---- load phase: one row per thread ----
    {
        const int l = tid;
        const size_t gl = (size_t)(c * CHNK + l);
        float csv = Acs[(size_t)blk * CHNK + l];
        scs[l] = csv;
        float dec = __expf(Acs[(size_t)blk * CHNK + 255] - csv);
        const uint4* gB = (const uint4*)(Bbf + gl * GN + grp * DSTATE);
        const uint4* gC = (const uint4*)(Cbf + gl * GN + grp * DSTATE);
#pragma unroll
        for (int j = 0; j < 16; ++j) {
            Q8 t; t.q = gB[j];
            *(uint4*)&sB[l * S_LB + j * 8] = t.q;
#pragma unroll
            for (int e = 0; e < 8; ++e) sBT[(j * 8 + e) * S_LT + l] = t.u[e];
            Q8 tc; tc.q = gC[j];
            *(uint4*)&sC[l * S_LB + j * 8] = tc.q;
        }
        const uint4* gX = (const uint4*)(xdt_bf + gl * DINNER + h * HD);
#pragma unroll
        for (int j = 0; j < 8; ++j) {
            Q8 t; t.q = gX[j];
#pragma unroll
            for (int e = 0; e < 8; ++e) {
                int p = j * 8 + e;
                sXdT[p * S_LT + l] = t.u[e];
                sXddT[p * S_LT + l] = f2bf(bf2f(t.u[e]) * dec);
            }
        }
    }
    __syncthreads();

    const int am = lane & 15;
    const int ak = (lane >> 4) * 8;
    const int bk = (lane >> 4) * 8;
    unsigned short* sGw = sG + wid * 16 * S_LG;

    // ---- Y_diag: Y = (L ∘ (C B^T)) xdt ; each wave: rows wid*32..+31 ----
    const int rm = wid * 32;
    AccF yacc[2][4] = {};
    for (int sb = 0; sb < CHNK; sb += 32) {
#pragma unroll
        for (int mt = 0; mt < 2; ++mt) {
            const int mrow = rm + mt * 16;
#pragma unroll
            for (int st2 = 0; st2 < 2; ++st2) {
                const int scol = sb + st2 * 16;
                AccF g = {};
#pragma unroll
                for (int kk = 0; kk < 4; ++kk) {
                    const unsigned short* ra = &sC[(mrow + am) * S_LB + kk * 32 + ak];
                    const unsigned short* rb = &sB[(scol + am) * S_LB + kk * 32 + bk];
                    g.v = wmma_bf(ld_frag(ra, ra + 16), ld_frag(rb, rb + 16), g.v);
                }
                // apply causal decay mask L[i,j] = exp(cs_i - cs_j), j<=i
                const int i0 = mrow + ((lane >> 4) << 3);
                const int j = scol + am;
                const float csj = scs[j];
#pragma unroll
                for (int r = 0; r < 8; ++r) {
                    const int i = i0 + r;
                    float gv = (j <= i) ? g.f[r] * __expf(scs[i] - csj) : 0.0f;
                    sGw[(((lane >> 4) << 3) + r) * S_LG + st2 * 16 + am] = f2bf(gv);
                }
            }
            // masked G tile (16 x 32) as A operand; accumulate against xdt^T
            const unsigned short* rg = &sGw[am * S_LG + ak];
            bf16x16 gf = ld_frag(rg, rg + 16);
#pragma unroll
            for (int pt = 0; pt < 4; ++pt) {
                const unsigned short* rx = &sXdT[(pt * 16 + am) * S_LT + sb + bk];
                yacc[mt][pt].v = wmma_bf(gf, ld_frag(rx, rx + 16), yacc[mt][pt].v);
            }
        }
    }
#pragma unroll
    for (int mt = 0; mt < 2; ++mt)
#pragma unroll
        for (int pt = 0; pt < 4; ++pt)
#pragma unroll
            for (int r = 0; r < 8; ++r) {
                int i = rm + mt * 16 + ((lane >> 4) << 3) + r;
                int p = pt * 16 + am;
                Ybuf[((size_t)(c * CHNK + i) * NH + h) * HD + p] = yacc[mt][pt].f[r];
            }

    // ---- states[p][n] = sum_l (xdt*decay)[l,p] * B[l,n] ----
    const int prow = (wid & 3) * 16;
    const int nc0 = (wid >> 2) * 64;
    AccF sacc[4] = {};
    for (int kb = 0; kb < CHNK; kb += 32) {
        const unsigned short* ra = &sXddT[(prow + am) * S_LT + kb + ak];
        bf16x16 af = ld_frag(ra, ra + 16);
#pragma unroll
        for (int nt = 0; nt < 4; ++nt) {
            const unsigned short* rb = &sBT[(nc0 + nt * 16 + am) * S_LT + kb + bk];
            sacc[nt].v = wmma_bf(af, ld_frag(rb, rb + 16), sacc[nt].v);
        }
    }
#pragma unroll
    for (int nt = 0; nt < 4; ++nt)
#pragma unroll
        for (int r = 0; r < 8; ++r) {
            int p = prow + ((lane >> 4) << 3) + r;
            int n = nc0 + nt * 16 + am;
            statesBuf[((size_t)blk * HD + p) * DSTATE + n] = sacc[nt].f[r];
        }
}

// ---------------- inter-chunk recurrence: init[c+1]=exp(dAsum[c])*init[c]+states[c] ----------------
__global__ void recurK(const float* __restrict__ statesBuf, const float* __restrict__ dAsum,
                       float* __restrict__ initBuf) {
    const int h = blockIdx.x, t = threadIdx.x;
    float st[32];
#pragma unroll
    for (int j = 0; j < 32; ++j) st[j] = 0.0f;
    for (int c = 0; c < NCH; ++c) {
        const size_t base = (size_t)(c * NH + h) * (HD * DSTATE);
        const float dec = __expf(dAsum[c * NH + h]);
#pragma unroll
        for (int j = 0; j < 32; ++j) initBuf[base + t + j * 256] = st[j];
#pragma unroll
        for (int j = 0; j < 32; ++j)
            st[j] = dec * st[j] + statesBuf[base + t + j * 256];
    }
}

// ---------------- Y_off = (C @ init_state) ∘ exp(cs) ; y_pre = Ydiag + Yoff + D*x ----------------
#define O_LB 136
__global__ void yoffK(const unsigned short* __restrict__ Cbf,
                      const float* __restrict__ initBuf, const float* __restrict__ Acs,
                      const float* __restrict__ Ybuf, const float* __restrict__ x_f32,
                      const float* __restrict__ Dp, float* __restrict__ yPre) {
    extern __shared__ char smem[];
    unsigned short* sC = (unsigned short*)smem;      // [256][O_LB]
    unsigned short* sS = sC + 256 * O_LB;            // [64][O_LB] state[p][n]
    float* scs = (float*)(sS + 64 * O_LB);           // [256]

    const int blk = blockIdx.x, c = blk >> 7, h = blk & 127;
    const int tid = threadIdx.x, lane = tid & 31, wid = tid >> 5;
    const int grp = h >> 4;
    {
        const int l = tid;
        const uint4* gC = (const uint4*)(Cbf + (size_t)(c * CHNK + l) * GN + grp * DSTATE);
#pragma unroll
        for (int j = 0; j < 16; ++j) *(uint4*)&sC[l * O_LB + j * 8] = gC[j];
        scs[l] = Acs[(size_t)blk * CHNK + l];
        for (int idx = tid; idx < HD * DSTATE; idx += 256) {
            int p = idx >> 7, n = idx & 127;
            sS[p * O_LB + n] = f2bf(initBuf[((size_t)blk * HD + p) * DSTATE + n]);
        }
    }
    __syncthreads();

    const int am = lane & 15, ak = (lane >> 4) * 8, bk = (lane >> 4) * 8;
    const int rm = wid * 32;
    AccF acc[2][4] = {};
#pragma unroll
    for (int kk = 0; kk < 4; ++kk) {
#pragma unroll
        for (int mt = 0; mt < 2; ++mt) {
            const unsigned short* ra = &sC[(rm + mt * 16 + am) * O_LB + kk * 32 + ak];
            bf16x16 af = ld_frag(ra, ra + 16);
#pragma unroll
            for (int pt = 0; pt < 4; ++pt) {
                const unsigned short* rb = &sS[(pt * 16 + am) * O_LB + kk * 32 + bk];
                acc[mt][pt].v = wmma_bf(af, ld_frag(rb, rb + 16), acc[mt][pt].v);
            }
        }
    }
    const float Dh = Dp[h];
#pragma unroll
    for (int mt = 0; mt < 2; ++mt)
#pragma unroll
        for (int pt = 0; pt < 4; ++pt)
#pragma unroll
            for (int r = 0; r < 8; ++r) {
                int i = rm + mt * 16 + ((lane >> 4) << 3) + r;
                int p = pt * 16 + am;
                size_t l = (size_t)(c * CHNK + i);
                float od = __expf(scs[i]);
                float v = acc[mt][pt].f[r] * od
                        + Ybuf[(l * NH + h) * HD + p]
                        + Dh * x_f32[l * DINNER + h * HD + p];
                yPre[l * DINNER + h * HD + p] = v;
            }
}

// ---------------- gated RMSNorm: y = y*silu(z); y *= rsqrt(mean(y^2)+eps)*w; -> bf16 ----------------
__global__ void normK(const float* __restrict__ yPre, const float* __restrict__ zx,
                      const float* __restrict__ nw, unsigned short* __restrict__ ybf) {
    const int l = blockIdx.x, t = threadIdx.x;
    __shared__ float red[256];
    float tv[32];
    float ss = 0.0f;
#pragma unroll
    for (int j = 0; j < 32; ++j) {
        int ii = t + j * 256;
        float g = siluf(zx[(size_t)l * DPROJ + ii]);
        float v = yPre[(size_t)l * DINNER + ii] * g;
        tv[j] = v;
        ss += v * v;
    }
    red[t] = ss;
    __syncthreads();
    for (int off = 128; off > 0; off >>= 1) {
        if (t < off) red[t] += red[t + off];
        __syncthreads();
    }
    float scale = rsqrtf(red[0] / (float)DINNER + 1e-5f);
#pragma unroll
    for (int j = 0; j < 32; ++j) {
        int ii = t + j * 256;
        ybf[(size_t)l * DINNER + ii] = f2bf(tv[j] * scale * nw[ii]);
    }
}

// ---------------- host-side orchestration ----------------
extern "C" void kernel_launch(void* const* d_in, const int* in_sizes, int n_in,
                              void* d_out, int out_size, void* d_ws, size_t ws_size,
                              hipStream_t stream) {
    (void)in_sizes; (void)n_in; (void)out_size; (void)ws_size;
    const float* hidden   = (const float*)d_in[0];
    const float* inproj_w = (const float*)d_in[1];
    const float* conv_w   = (const float*)d_in[2];
    const float* conv_b   = (const float*)d_in[3];
    const float* dt_bias  = (const float*)d_in[4];
    const float* A_log    = (const float*)d_in[5];
    const float* Dp       = (const float*)d_in[6];
    const float* norm_w   = (const float*)d_in[7];
    const float* outproj_w= (const float*)d_in[8];
    float* out = (float*)d_out;

    char* ws = (char*)d_ws;
    size_t off = 0;
    auto take = [&](size_t bytes) -> void* {
        void* p = ws + off;
        off += (bytes + 255) & ~(size_t)255;
        return p;
    };
    unsigned short* Xbf    = (unsigned short*)take((size_t)SEQ * DMODEL * 2);
    unsigned short* WinBf  = (unsigned short*)take((size_t)DPROJ * DMODEL * 2);
    unsigned short* WoutBf = (unsigned short*)take((size_t)DMODEL * DINNER * 2);
    float* zx      = (float*)take((size_t)SEQ * DPROJ * 4);
    float* dt      = (float*)take((size_t)SEQ * NH * 4);
    float* dA      = (float*)take((size_t)SEQ * NH * 4);
    float* Acs     = (float*)take((size_t)NCH * NH * CHNK * 4);
    float* dAsum   = (float*)take((size_t)NCH * NH * 4);
    float* x_f32   = (float*)take((size_t)SEQ * DINNER * 4);
    unsigned short* xdt_bf = (unsigned short*)take((size_t)SEQ * DINNER * 2);
    unsigned short* Bbf    = (unsigned short*)take((size_t)SEQ * GN * 2);
    unsigned short* Cbf    = (unsigned short*)take((size_t)SEQ * GN * 2);
    float* Ybuf    = (float*)take((size_t)SEQ * DINNER * 4);
    float* states  = (float*)take((size_t)NCH * NH * HD * DSTATE * 4);
    float* initSt  = (float*)take((size_t)NCH * NH * HD * DSTATE * 4);
    float* yPre    = (float*)take((size_t)SEQ * DINNER * 4);
    unsigned short* ybf = (unsigned short*)take((size_t)SEQ * DINNER * 2);

    // 1. casts (weights become bf16 -> in_proj fits in 192MB L2)
    castK<<<4096, 256, 0, stream>>>(hidden, Xbf, (size_t)SEQ * DMODEL);
    castK<<<8192, 256, 0, stream>>>(inproj_w, WinBf, (size_t)DPROJ * DMODEL);
    castK<<<8192, 256, 0, stream>>>(outproj_w, WoutBf, (size_t)DMODEL * DINNER);

    // 2. in_proj GEMM: zx[2048,18560] = X @ Win^T
    gemmK<<<dim3(DPROJ / 128, SEQ / 128), 256, 0, stream>>>(
        Xbf, WinBf, zx, nullptr, SEQ, DPROJ, DMODEL);

    // 3. dt / dA
    dtK<<<(SEQ * NH) / 256, 256, 0, stream>>>(zx, dt_bias, A_log, dt, dA);

    // 4. conv + silu + split + xdt
    convK_<<<(SEQ * CONVDIM) / 256, 256, 0, stream>>>(zx, conv_w, conv_b, dt,
                                                      x_f32, xdt_bf, Bbf, Cbf);

    // 5. per-(chunk,head) cumsum
    cumsumK<<<NCH * NH, 256, 0, stream>>>(dA, Acs, dAsum);

    // 6. chunk scan (WMMA): Y_diag + per-chunk states
    const size_t scan_lds =
        (size_t)(256 * S_LB + 256 * S_LB + 128 * S_LT + 64 * S_LT + 64 * S_LT +
                 8 * 16 * S_LG) * 2 + 256 * 4;
    scanK<<<NCH * NH, 256, scan_lds, stream>>>(Bbf, Cbf, xdt_bf, Acs, Ybuf, states);

    // 7. inter-chunk recurrence
    recurK<<<NH, 256, 0, stream>>>(states, dAsum, initSt);

    // 8. Y_off (WMMA) + D*x
    const size_t yoff_lds = (size_t)(256 * O_LB + 64 * O_LB) * 2 + 256 * 4;
    yoffK<<<NCH * NH, 256, yoff_lds, stream>>>(Cbf, initSt, Acs, Ybuf, x_f32, Dp, yPre);

    // 9. gating + RMSNorm -> bf16
    normK<<<SEQ, 256, 0, stream>>>(yPre, zx, norm_w, ybf);

    // 10. out_proj GEMM + residual: out = ybf @ Wout^T + hidden
    gemmK<<<dim3(DMODEL / 128, SEQ / 128), 256, 0, stream>>>(
        ybf, WoutBf, out, hidden, SEQ, DMODEL, DINNER);
}
